// SIFA_66958540145155
// MI455X (gfx1250) — matrix operand order repaired and checked by reading, not compiled, not run on hardware
//
#include <hip/hip_runtime.h>
#include <hip/hip_bf16.h>
#include <math.h>

typedef __attribute__((ext_vector_type(16))) __bf16 bf16x16;
typedef __attribute__((ext_vector_type(8)))  __bf16 bf16x8;
typedef __attribute__((ext_vector_type(8)))  float  f32x8;

#define BATCH 8
#define C_IN  64
#define HH    128
#define WWID  128
#define HWP   (HH*WWID)        // 16384
#define CE    (C_IN*9)         // 576
#define OUP   128
#define EPSV  1e-5f

// fast sigmoid: v_exp_f32 + v_rcp_f32 (no div_fixup chain)
__device__ __forceinline__ float sigmoidf_(float x){
    return __builtin_amdgcn_rcpf(1.0f + __expf(-x));
}

// ---------------- Kernel 0: prep ----------------
// bf16 weight matrix W2[128][576] (conv_w OIHW is already that layout),
// BN scale/shift for gen-BN and out-BN, zero the stats accumulator.
__global__ void sifa_prep(const float* __restrict__ gen_g, const float* __restrict__ gen_b,
                          const float* __restrict__ gen_m, const float* __restrict__ gen_v,
                          const float* __restrict__ conv_w,
                          const float* __restrict__ conv_g, const float* __restrict__ conv_b,
                          const float* __restrict__ conv_m, const float* __restrict__ conv_v,
                          __bf16* __restrict__ W2, float* __restrict__ genScale,
                          float* __restrict__ genShift, float* __restrict__ zScale,
                          float* __restrict__ zShift, float* __restrict__ stats)
{
    int tid = blockIdx.x*blockDim.x + threadIdx.x;
    if (tid < OUP*CE) W2[tid] = (__bf16)conv_w[tid];
    if (tid < CE) {
        float s = gen_g[tid] * rsqrtf(gen_v[tid] + EPSV);
        genScale[tid] = s;
        genShift[tid] = gen_b[tid] - gen_m[tid]*s;
    }
    if (tid < OUP) {
        float s = conv_g[tid] * rsqrtf(conv_v[tid] + EPSV);
        zScale[tid] = s;
        zShift[tid] = conv_b[tid] - conv_m[tid]*s;
    }
    if (tid < BATCH*C_IN*2) stats[tid] = 0.0f;
}

// ---------------- Kernel 1: depthwise conv + BN + ReLU + stats ----------------
// grid = B*C*64 blocks of 256; block handles one (b,c) and 256 pixels.
// Writes v in pixel-major layout: vws[(b*HWP+hw)*CE + c*9 + j] (bf16).
__global__ __launch_bounds__(256) void sifa_gen(
        const float* __restrict__ x, const float* __restrict__ genw,
        const float* __restrict__ genScale, const float* __restrict__ genShift,
        __bf16* __restrict__ vws, float* __restrict__ stats)
{
    __shared__ float wsm[81];
    __shared__ float scl[9], sft[9];
    __shared__ float redS[8], redQ[8];

    int tid   = threadIdx.x;
    int bc    = blockIdx.x >> 6;      // (b*64 + c)
    int chunk = blockIdx.x & 63;
    int b = bc >> 6;
    int c = bc & 63;

    if (tid < 81)      wsm[tid]      = genw[c*81 + tid];
    else if (tid < 90) scl[tid-81]   = genScale[c*9 + (tid-81)];
    else if (tid < 99) sft[tid-90]   = genShift[c*9 + (tid-90)];
    __syncthreads();

    int hw = chunk*256 + tid;
    int h  = hw >> 7;
    int w  = hw & 127;
    const float* xc = x + (size_t)bc * HWP;

    float win[9];
    #pragma unroll
    for (int dh = -1; dh <= 1; ++dh) {
        #pragma unroll
        for (int dw = -1; dw <= 1; ++dw) {
            int hh = h + dh, ww = w + dw;
            bool ok = (hh >= 0) & (hh < HH) & (ww >= 0) & (ww < WWID);
            win[(dh+1)*3 + (dw+1)] = ok ? xc[hh*WWID + ww] : 0.0f;
        }
    }

    __bf16* dst = vws + (size_t)(b*HWP + hw)*CE + c*9;
    float sum = 0.0f, sq = 0.0f;
    #pragma unroll
    for (int j = 0; j < 9; ++j) {
        float acc = 0.0f;
        #pragma unroll
        for (int t = 0; t < 9; ++t) acc = fmaf(win[t], wsm[j*9 + t], acc);
        float val = fmaxf(0.0f, fmaf(acc, scl[j], sft[j]));
        sum += val; sq = fmaf(val, val, sq);
        dst[j] = (__bf16)val;
    }

    // wave32 shuffle reduce, then cross-wave via LDS, then 2 atomics per block
    #pragma unroll
    for (int off = 16; off > 0; off >>= 1) {
        sum += __shfl_down(sum, off, 32);
        sq  += __shfl_down(sq,  off, 32);
    }
    int wave = tid >> 5, lane = tid & 31;
    if (lane == 0) { redS[wave] = sum; redQ[wave] = sq; }
    __syncthreads();
    if (tid == 0) {
        float ts = 0.0f, tq = 0.0f;
        #pragma unroll
        for (int i = 0; i < 8; ++i) { ts += redS[i]; tq += redQ[i]; }
        atomicAdd(&stats[2*bc + 0], ts);
        atomicAdd(&stats[2*bc + 1], tq);
    }
}

// ---------------- Kernel 2: finalize gate params ----------------
__global__ void sifa_stats(const float* __restrict__ stats,
                           const float* __restrict__ cw, const float* __restrict__ cb,
                           const float* __restrict__ sw, const float* __restrict__ sb,
                           const float* __restrict__ gnw, const float* __restrict__ gnb,
                           const float* __restrict__ wadd, float* __restrict__ gate)
{
    int tid = threadIdx.x;
    if (tid >= BATCH*C_IN) return;
    int c  = tid & 63;
    int cg = c & 7;
    const float inv = 1.0f / (9.0f * (float)HWP);
    float S1 = stats[2*tid], S2 = stats[2*tid+1];
    float mean = S1 * inv;
    float var  = fmaxf(0.0f, S2*inv - mean*mean);
    float rstd = rsqrtf(var + EPSV);
    float wa   = wadd[0];
    float sch  = wa * sigmoidf_(cw[cg]*mean + cb[cg]);
    float A    = sw[cg]*gnw[cg]*rstd;
    float OFF  = sw[cg]*(gnb[cg] - mean*gnw[cg]*rstd) + sb[cg];
    gate[4*tid+0] = sch;
    gate[4*tid+1] = A;
    gate[4*tid+2] = OFF;
    gate[4*tid+3] = 1.0f - wa;
}

// ---------------- Kernel 3: fused gate + WMMA GEMM + BN + SiLU ----------------
// D[128 x N] = W2[128 x 576] * gated_v[576 x N], N = B*HWP.
// Block: 256 thr (8 waves), tile 128(M) x 128(N); wave w -> 16-col strip,
// full M via 8 accumulators of v_wmma_f32_16x16x32_bf16.
__global__ __launch_bounds__(256) void sifa_gemm(
        const __bf16* __restrict__ vws, const __bf16* __restrict__ W2,
        const float* __restrict__ gate,
        const float* __restrict__ zScale, const float* __restrict__ zShift,
        float* __restrict__ out)
{
    __shared__ float gsch[C_IN], gA[C_IN], gOFF[C_IN];
    __shared__ float zs[OUP], zf[OUP];
    __shared__ float wbS;

    int tid    = threadIdx.x;
    int wave   = tid >> 5;
    int lane   = tid & 31;
    int ln     = lane & 15;
    int hiHalf = lane >> 4;       // 0 or 1

    int colBlock = blockIdx.x * 128;
    int b = colBlock >> 14;       // 128 | 16384 -> block never straddles batches

    if (tid < C_IN) {
        gsch[tid] = gate[4*(b*C_IN + tid) + 0];
        gA[tid]   = gate[4*(b*C_IN + tid) + 1];
        gOFF[tid] = gate[4*(b*C_IN + tid) + 2];
        if (tid == 0) wbS = gate[3];
    }
    if (tid < OUP) { zs[tid] = zScale[tid]; zf[tid] = zShift[tid]; }
    __syncthreads();
    float wb = wbS;

    int col = colBlock + wave*16 + ln;
    const __bf16* vcol = vws + (size_t)col * CE;

    f32x8 acc[8];
    #pragma unroll
    for (int i = 0; i < 8; ++i)
        #pragma unroll
        for (int r = 0; r < 8; ++r) acc[i][r] = 0.0f;

    const int kB = hiHalf * 16;   // B frag: K rows (lane<16 ? 0..15 : 16..31)
    const int kA = hiHalf * 16;   // A frag: K rows (lane<16 ? 0..15 : 16..31)

    for (int kc = 0; kc < CE; kc += 32) {
        // --- B fragment: 16 consecutive ce from this column, gated in-register ---
        bf16x16 braw = *(const bf16x16*)(vcol + kc + kB);
        bf16x16 bfrag;
        #pragma unroll
        for (int i = 0; i < 16; ++i) {
            int ce = kc + kB + i;
            int c  = ce / 9;
            float f = (float)braw[i];
            float g = f * (gsch[c] + wb * sigmoidf_(fmaf(gA[c], f, gOFF[c])));
            bfrag[i] = (__bf16)g;
        }
        // --- 8 M-tiles of A, 8 WMMAs ---
        #pragma unroll
        for (int mt = 0; mt < 8; ++mt) {
            const __bf16* wrow = W2 + (size_t)(mt*16 + ln)*CE + kc + kA;
            union { bf16x16 v; bf16x8 h[2]; } au;
            au.h[0] = *(const bf16x8*)(wrow);
            au.h[1] = *(const bf16x8*)(wrow + 8);
            acc[mt] = __builtin_amdgcn_wmma_f32_16x16x32_bf16(
                false, au.v, false, bfrag, (short)0, acc[mt], false, false);
        }
    }

    // --- epilogue: BN + SiLU, scatter to NCHW [B,128,16384] ---
    int pixel = col & (HWP - 1);
    #pragma unroll
    for (int mt = 0; mt < 8; ++mt) {
        #pragma unroll
        for (int r = 0; r < 8; ++r) {
            int M = mt*16 + r + hiHalf*8;
            float z = fmaf(acc[mt][r], zs[M], zf[M]);
            out[((size_t)(b*OUP + M) << 14) + pixel] = z * sigmoidf_(z);
        }
    }
}

extern "C" void kernel_launch(void* const* d_in, const int* in_sizes, int n_in,
                              void* d_out, int out_size, void* d_ws, size_t ws_size,
                              hipStream_t stream) {
    (void)in_sizes; (void)n_in; (void)out_size; (void)ws_size;
    const float* x        = (const float*)d_in[0];
    const float* gen_w    = (const float*)d_in[1];
    const float* gen_bn_g = (const float*)d_in[2];
    const float* gen_bn_b = (const float*)d_in[3];
    const float* gen_bn_m = (const float*)d_in[4];
    const float* gen_bn_v = (const float*)d_in[5];
    const float* gn_w     = (const float*)d_in[6];
    const float* gn_b     = (const float*)d_in[7];
    const float* cweight  = (const float*)d_in[8];
    const float* cbias    = (const float*)d_in[9];
    const float* sweight  = (const float*)d_in[10];
    const float* sbias    = (const float*)d_in[11];
    const float* wadd     = (const float*)d_in[12];
    const float* conv_w   = (const float*)d_in[13];
    const float* conv_bn_g= (const float*)d_in[14];
    const float* conv_bn_b= (const float*)d_in[15];
    const float* conv_bn_m= (const float*)d_in[16];
    const float* conv_bn_v= (const float*)d_in[17];
    float* out = (float*)d_out;

    char* ws = (char*)d_ws;
    size_t off = 0;
    __bf16* vws      = (__bf16*)(ws + off); off += (size_t)BATCH*HWP*CE*sizeof(__bf16); // ~151 MB
    __bf16* W2       = (__bf16*)(ws + off); off += (size_t)OUP*CE*sizeof(__bf16);
    float*  genScale = (float*) (ws + off); off += CE*sizeof(float);
    float*  genShift = (float*) (ws + off); off += CE*sizeof(float);
    float*  zScale   = (float*) (ws + off); off += OUP*sizeof(float);
    float*  zShift   = (float*) (ws + off); off += OUP*sizeof(float);
    float*  stats    = (float*) (ws + off); off += (size_t)BATCH*C_IN*2*sizeof(float);
    float*  gateT    = (float*) (ws + off); off += (size_t)BATCH*C_IN*4*sizeof(float);

    sifa_prep<<<288, 256, 0, stream>>>(gen_bn_g, gen_bn_b, gen_bn_m, gen_bn_v,
                                       conv_w, conv_bn_g, conv_bn_b, conv_bn_m, conv_bn_v,
                                       W2, genScale, genShift, zScale, zShift, stats);
    sifa_gen<<<BATCH*C_IN*64, 256, 0, stream>>>(x, gen_w, genScale, genShift, vws, stats);
    sifa_stats<<<1, 512, 0, stream>>>(stats, cweight, cbias, sweight, sbias,
                                      gn_w, gn_b, wadd, gateT);
    sifa_gemm<<<(BATCH*HWP)/128, 256, 0, stream>>>(vws, W2, gateT, zScale, zShift, out);
}
